// FAVORPlusAttention_67568425501053
// MI455X (gfx1250) — matrix operand order repaired and checked by reading, not compiled, not run on hardware
//
#include <hip/hip_runtime.h>
#include <hip/hip_bf16.h>

typedef __bf16 bf16_t;
typedef __attribute__((ext_vector_type(16))) __bf16 v16bf;
typedef __attribute__((ext_vector_type(8)))  __bf16 v8bf;
typedef __attribute__((ext_vector_type(8)))  float  v8f;
typedef __attribute__((ext_vector_type(4)))  float  v4f;
typedef __attribute__((ext_vector_type(4)))  unsigned int uint32x4;
typedef __attribute__((ext_vector_type(8)))  int int32x8;
typedef __attribute__((ext_vector_type(4)))  int int32x4;

__device__ __forceinline__ bf16_t f2bf(float x) { return (bf16_t)x; }
__device__ __forceinline__ float  bf2f(bf16_t x){ return (float)x; }

__device__ __forceinline__ v8f zero8() {
  v8f z = {0.f,0.f,0.f,0.f,0.f,0.f,0.f,0.f};
  return z;
}

__device__ __forceinline__ v8f wmma_bf16(v16bf a, v16bf b, v8f c) {
  return __builtin_amdgcn_wmma_f32_16x16x32_bf16(
      /*neg_a=*/false, a, /*neg_b=*/false, b,
      /*c_mod=*/(short)0, c, /*reuse_a=*/false, /*reuse_b=*/false);
}

// A fragment (16x32 bf16, MxK) from [row][k] tile, ldk = row stride (elems).
// Lane l: row = l&15; k runs = {h*8+0..7, 16+h*8+0..7}.
__device__ __forceinline__ v16bf load_fragA(const bf16_t* tile, int ldk, int lane) {
  int r = lane & 15, half = lane >> 4;
  const bf16_t* p = tile + (size_t)r * ldk + half * 8;
  v8bf lo = *(const v8bf*)(p);
  v8bf hi = *(const v8bf*)(p + 16);
  v16bf a;
#pragma unroll
  for (int i = 0; i < 8; i++) { a[i] = lo[i]; a[i + 8] = hi[i]; }
  return a;
}

// B fragment (32x16 bf16, KxN) from a TRANSPOSED tile Bt[col][k].
// Lane l: col = l&15; k runs = {h*8+0..7, 16+h*8+0..7}.
__device__ __forceinline__ v16bf load_fragB(const bf16_t* tileT, int ldk, int lane) {
  int c = lane & 15, half = lane >> 4;
  const bf16_t* p = tileT + (size_t)c * ldk + half * 8;
  v8bf lo = *(const v8bf*)(p);
  v8bf hi = *(const v8bf*)(p + 16);
  v16bf b;
#pragma unroll
  for (int i = 0; i < 8; i++) { b[i] = lo[i]; b[i + 8] = hi[i]; }
  return b;
}

// A fragment of M^T from a ROW-MAJOR tile T[n][w] (k dim = n rows).
// Lane l: A-row r = l&15 -> column (col0 + r) of T; k = n per A layout.
__device__ __forceinline__ v16bf load_fragA_colgather(const bf16_t* tile, int ldw,
                                                      int col0, int lane) {
  int r = lane & 15, base8 = (lane >> 4) * 8;
  v16bf a;
#pragma unroll
  for (int i = 0; i < 8; i++) {
    a[i]     = tile[(size_t)(base8 + i) * ldw + col0 + r];
    a[i + 8] = tile[(size_t)(16 + base8 + i) * ldw + col0 + r];
  }
  return a;
}

// B fragment from ROW-MAJOR tile T[n][w]: lane col = col0 + (l&15),
// k = {(l>>4)*8 + 0..7, 16 + (l>>4)*8 + 0..7} over n rows.
__device__ __forceinline__ v16bf load_fragB_colgather(const bf16_t* tile, int ldw,
                                                      int col0, int lane) {
  int c = col0 + (lane & 15), k0 = (lane >> 4) * 8;
  v16bf b;
#pragma unroll
  for (int e = 0; e < 16; e++)
    b[e] = tile[(size_t)(k0 + e + (e >= 8 ? 8 : 0)) * ldw + c];
  return b;
}

__device__ __forceinline__ void load4(const float* p, float f[4]) {
  v4f v = *(const v4f*)p;
  f[0] = v[0]; f[1] = v[1]; f[2] = v[2]; f[3] = v[3];
}
__device__ __forceinline__ void load4(const bf16_t* p, float f[4]) {
#pragma unroll
  for (int q = 0; q < 4; q++) f[q] = bf2f(p[q]);
}

// ---------------------------------------------------------------------------
// TDM: DMA a 2-D bf16 tile (tile_h rows x tile_w elems, row stride
// row_stride elems) from global memory into LDS at lds_byte_off.
// Descriptor per CDNA5 ISA 8.3/8.4 (group0 + group1, groups 2/3 zero = 2D).
// ---------------------------------------------------------------------------
__device__ __forceinline__ void tdm_load_2d_bf16(
    unsigned int lds_byte_off, const void* gptr,
    unsigned int tile_w, unsigned int tile_h,
    unsigned int row_stride, unsigned int tensor_w, unsigned int tensor_h) {
  unsigned long long ga = (unsigned long long)gptr;
  uint32x4 g0 = {
      1u,                                   // count=1 (valid), user mode
      lds_byte_off,                         // lds_addr (bytes)
      (unsigned int)ga,                     // global_addr[31:0]
      (unsigned int)((ga >> 32) & 0x01FFFFFFu) | (2u << 30)  // addr[56:32] | type=2
  };
  int32x8 g1 = {
      (int)(1u << 16),                                          // data_size=1 (2B)
      (int)((tensor_w & 0xFFFFu) << 16),                        // [63:48]=dim0 lo
      (int)((tensor_w >> 16) | ((tensor_h & 0xFFFFu) << 16)),   // dim0 hi | dim1 lo
      (int)((tensor_h >> 16) | (tile_w << 16)),                 // dim1 hi | tile_dim0
      (int)(tile_h & 0xFFFFu),                                  // tile_dim1 (tile_dim2=0)
      (int)row_stride,                                          // dim0_stride[31:0]
      0,                                                        // stride hi | dim1_stride lo
      0
  };
  int32x4 gz = {0, 0, 0, 0};
#if __clang_major__ >= 23
  int32x8 gz8 = {0, 0, 0, 0, 0, 0, 0, 0};
  __builtin_amdgcn_tensor_load_to_lds(g0, g1, gz, gz, gz8, 0);
#else
  __builtin_amdgcn_tensor_load_to_lds(g0, g1, gz, gz, 0);
#endif
}

// ---------------------------------------------------------------------------
// Generic bf16-WMMA GEMM: C[M,Nn] = A[M,K] * Bw[K,Nn] + bias[Nn]
// Block tile 128x128, BK=32. 8 waves: wave w covers rows (w&3)*32..+31,
// cols (w>>2)*64..+63 -> 2x4 grid of 16x16 WMMA tiles.
// ---------------------------------------------------------------------------
template <typename AT, typename OT>
__global__ __launch_bounds__(256) void gemm_bf16_kernel(
    const AT* __restrict__ A, const float* __restrict__ Bw,
    const float* __restrict__ bias, OT* __restrict__ C,
    int M, int K, int Nn) {
  __shared__ bf16_t As[128 * 32];   // [row][k]
  __shared__ bf16_t Bt[128 * 32];   // [col][k]  (transposed stage)

  const int t = threadIdx.x;
  const int lane = t & 31;
  const int w = t >> 5;
  const int m0 = blockIdx.x * 128;
  const int n0 = blockIdx.y * 128;
  const int wm = (w & 3) * 32;
  const int wn = (w >> 2) * 64;

  v8f acc[2][4];
#pragma unroll
  for (int i = 0; i < 2; i++)
#pragma unroll
    for (int j = 0; j < 4; j++) acc[i][j] = zero8();

  for (int k0 = 0; k0 < K; k0 += 32) {
    // stage A tile: 128 rows x 32 k (4096 elems, 16/thread)
#pragma unroll
    for (int i = 0; i < 4; i++) {
      int j = t * 4 + i;
      int row = j >> 3;
      int kc = (j & 7) * 4;
      float f[4];
      load4(A + (size_t)(m0 + row) * K + k0 + kc, f);
#pragma unroll
      for (int q = 0; q < 4; q++) As[row * 32 + kc + q] = f2bf(f[q]);
    }
    // stage B tile transposed: 32 k x 128 n -> Bt[n][k]
#pragma unroll
    for (int i = 0; i < 4; i++) {
      int j = t * 4 + i;
      int kr = j >> 5;
      int nc = (j & 31) * 4;
      float f[4];
      load4(Bw + (size_t)(k0 + kr) * Nn + n0 + nc, f);
#pragma unroll
      for (int q = 0; q < 4; q++) Bt[(nc + q) * 32 + kr] = f2bf(f[q]);
    }
    __syncthreads();

    v16bf afrag[2], bfrag[4];
#pragma unroll
    for (int mt = 0; mt < 2; mt++)
      afrag[mt] = load_fragA(&As[(wm + mt * 16) * 32], 32, lane);
#pragma unroll
    for (int nt = 0; nt < 4; nt++)
      bfrag[nt] = load_fragB(&Bt[(wn + nt * 16) * 32], 32, lane);
#pragma unroll
    for (int mt = 0; mt < 2; mt++)
#pragma unroll
      for (int nt = 0; nt < 4; nt++)
        acc[mt][nt] = wmma_bf16(afrag[mt], bfrag[nt], acc[mt][nt]);
    __syncthreads();
  }

#pragma unroll
  for (int mt = 0; mt < 2; mt++) {
#pragma unroll
    for (int nt = 0; nt < 4; nt++) {
      int col = n0 + wn + nt * 16 + (lane & 15);
      float bv = bias[col];
#pragma unroll
      for (int e = 0; e < 8; e++) {
        int row = m0 + wm + mt * 16 + (lane >> 4) * 8 + e;
        float val = acc[mt][nt][e] + bv;
        if constexpr (__is_same(OT, float))
          C[(size_t)row * Nn + col] = val;
        else
          C[(size_t)row * Nn + col] = f2bf(val);
      }
    }
  }
}

// ---------------------------------------------------------------------------
// Feature map: q'/k' = 0.125 * exp(data @ P - 0.5*|data|^2 + 1e-6)
// grid = (N/128, 2 (q|k), B*H); block = 256 (8 waves, 16 rows each).
// ---------------------------------------------------------------------------
__global__ __launch_bounds__(256) void feature_kernel(
    const bf16_t* __restrict__ qkv, const float* __restrict__ proj,
    bf16_t* __restrict__ qp, bf16_t* __restrict__ kp) {
  __shared__ bf16_t Pt[64 * 64];    // [m][k] transposed projection

  const int t = threadIdx.x;
  const int lane = t & 31;
  const int w = t >> 5;
  const int bh = blockIdx.z;
  const int b = bh >> 4, h = bh & 15;
  const int qk = blockIdx.y;
  const int nbase = blockIdx.x * 128 + w * 16;

#pragma unroll
  for (int i = 0; i < 16; i++) {
    int j = t * 16 + i;
    int k = j >> 6, m = j & 63;
    Pt[m * 64 + k] = f2bf(proj[k * 64 + m]);
  }
  __syncthreads();

  const int r = lane & 15, half = lane >> 4, base8 = half * 8;
  const bf16_t* abase =
      qkv + ((size_t)(b * 4096 + nbase)) * 3072 + qk * 1024 + h * 64;
  const bf16_t* ap = abase + (size_t)r * 3072;

  v16bf afrag[2];
  float sq = 0.f;
#pragma unroll
  for (int ks = 0; ks < 2; ks++) {
    v8bf lo = *(const v8bf*)(ap + ks * 32 + base8);
    v8bf hi = *(const v8bf*)(ap + ks * 32 + 16 + base8);
    v16bf a;
#pragma unroll
    for (int i = 0; i < 8; i++) {
      a[i] = lo[i];     float f0 = bf2f(lo[i]); sq += f0 * f0;
      a[i + 8] = hi[i]; float f1 = bf2f(hi[i]); sq += f1 * f1;
    }
    afrag[ks] = a;
  }
  sq += __shfl_xor(sq, 16, 32);
  float norm = 0.5f * sq;           // lane l holds norm of row (l&15)

  v8f acc[4];
#pragma unroll
  for (int mt = 0; mt < 4; mt++) acc[mt] = zero8();
#pragma unroll
  for (int mt = 0; mt < 4; mt++)
#pragma unroll
    for (int ks = 0; ks < 2; ks++) {
      v16bf bb = load_fragB(&Pt[(mt * 16) * 64 + ks * 32], 64, lane);
      acc[mt] = wmma_bf16(afrag[ks], bb, acc[mt]);
    }

  bf16_t* dst = (qk ? kp : qp) + ((size_t)bh * 4096 + nbase) * 64;
#pragma unroll
  for (int mt = 0; mt < 4; mt++) {
#pragma unroll
    for (int e = 0; e < 8; e++) {
      int row = half * 8 + e;
      float nrm = __shfl(norm, row, 32);
      float val = 0.125f * __expf(acc[mt][e] - nrm + 1e-6f);
      dst[(size_t)row * 64 + mt * 16 + r] = f2bf(val);
    }
  }
}

// ---------------------------------------------------------------------------
// kv[d][m] = sum_n v[n,d]*k'[n,m];  ksum[m] = sum_n k'[n,m].  One block/head.
// Chunks of 32 rows are staged into LDS via the Tensor Data Mover (TDM);
// fragments are gathered from the row-major tiles.
// ---------------------------------------------------------------------------
__global__ __launch_bounds__(256) void kv_kernel(
    const bf16_t* __restrict__ qkv, const bf16_t* __restrict__ kp,
    float* __restrict__ kv, float* __restrict__ ksum) {
  __shared__ bf16_t Vrm[32 * 64];   // [n][d] row-major V chunk (TDM dest)
  __shared__ bf16_t Krm[32 * 64];   // [n][m] row-major K' chunk (TDM dest)
  __shared__ float ksum_s[64];

  const int t = threadIdx.x;
  const int lane = t & 31;
  const int w = t >> 5;
  const int bh = blockIdx.x;
  const int b = bh >> 4, h = bh & 15;

  if (t < 64) ksum_s[t] = 0.f;

  v8f acc[2];
  acc[0] = zero8(); acc[1] = zero8();

  const bf16_t* vbase = qkv + ((size_t)b * 4096) * 3072 + 2048 + h * 64;
  const bf16_t* kbase = kp + ((size_t)bh * 4096) * 64;
  const unsigned int vrm_off = (unsigned int)(size_t)(void*)Vrm;
  const unsigned int krm_off = (unsigned int)(size_t)(void*)Krm;
  __syncthreads();

  for (int s = 0; s < 128; s++) {
    int nb = s * 32;
    if (t < 32) {
      // TDM: 32 rows x 64 bf16, strided (V) / contiguous (K')
      tdm_load_2d_bf16(vrm_off, vbase + (size_t)nb * 3072,
                       /*tile_w=*/64, /*tile_h=*/32,
                       /*row_stride=*/3072, /*tensor_w=*/64, /*tensor_h=*/32);
      tdm_load_2d_bf16(krm_off, kbase + (size_t)nb * 64,
                       /*tile_w=*/64, /*tile_h=*/32,
                       /*row_stride=*/64, /*tensor_w=*/64, /*tensor_h=*/32);
      __builtin_amdgcn_s_wait_tensorcnt(0);
    }
    __syncthreads();

#pragma unroll
    for (int u = 0; u < 2; u++) {
      int tid = w * 2 + u;
      int dt = tid >> 2, mt = tid & 3;
      v16bf a = load_fragA_colgather(Vrm, 64, dt * 16, lane);   // rows d, k=n
      v16bf bb = load_fragB_colgather(Krm, 64, mt * 16, lane);  // cols m, k=n
      acc[u] = wmma_bf16(a, bb, acc[u]);
    }
    if (t < 64) {
      float s2 = 0.f;
#pragma unroll 8
      for (int n = 0; n < 32; n++) s2 += bf2f(Krm[n * 64 + t]);
      ksum_s[t] += s2;
    }
    __syncthreads();
  }

#pragma unroll
  for (int u = 0; u < 2; u++) {
    int tid = w * 2 + u;
    int dt = tid >> 2, mt = tid & 3;
#pragma unroll
    for (int e = 0; e < 8; e++) {
      int d = dt * 16 + (lane >> 4) * 8 + e;
      int m = mt * 16 + (lane & 15);
      kv[((size_t)bh * 64 + d) * 64 + m] = acc[u][e];
    }
  }
  if (t < 64) ksum[(size_t)bh * 64 + t] = ksum_s[t];
}

// ---------------------------------------------------------------------------
// out[n,d] = (q'[n,:] . kv[d,:]) / (q'[n,:] . ksum + 1e-6) -> att bf16 [B,N,D]
// ---------------------------------------------------------------------------
__global__ __launch_bounds__(256) void out_kernel(
    const bf16_t* __restrict__ qp, const float* __restrict__ kv,
    const float* __restrict__ ksum, bf16_t* __restrict__ att) {
  __shared__ bf16_t Kv[64 * 64];    // [d][m]
  __shared__ float ks_s[64];

  const int t = threadIdx.x;
  const int lane = t & 31;
  const int w = t >> 5;
  const int bh = blockIdx.y;
  const int b = bh >> 4, h = bh & 15;
  const int nbase = blockIdx.x * 128 + w * 16;

#pragma unroll
  for (int i = 0; i < 16; i++) {
    int j = t * 16 + i;
    int d = j >> 6, m = j & 63;
    Kv[d * 64 + m] = f2bf(kv[((size_t)bh * 64 + d) * 64 + m]);
  }
  if (t < 64) ks_s[t] = ksum[(size_t)bh * 64 + t];
  __syncthreads();

  const int r = lane & 15, half = lane >> 4, base8 = half * 8;
  const bf16_t* ap = qp + ((size_t)bh * 4096 + nbase) * 64 + (size_t)r * 64;

  v16bf afrag[2];
  float dot = 0.f;
#pragma unroll
  for (int ks = 0; ks < 2; ks++) {
    v8bf lo = *(const v8bf*)(ap + ks * 32 + base8);
    v8bf hi = *(const v8bf*)(ap + ks * 32 + 16 + base8);
    v16bf a;
#pragma unroll
    for (int i = 0; i < 8; i++) {
      a[i] = lo[i];
      dot += bf2f(lo[i]) * ks_s[ks * 32 + base8 + i];
      a[i + 8] = hi[i];
      dot += bf2f(hi[i]) * ks_s[ks * 32 + 16 + base8 + i];
    }
    afrag[ks] = a;
  }
  dot += __shfl_xor(dot, 16, 32);

  v8f acc[4];
#pragma unroll
  for (int dt = 0; dt < 4; dt++) acc[dt] = zero8();
#pragma unroll
  for (int dt = 0; dt < 4; dt++)
#pragma unroll
    for (int ks = 0; ks < 2; ks++) {
      v16bf bb = load_fragB(&Kv[(dt * 16) * 64 + ks * 32], 64, lane);
      acc[dt] = wmma_bf16(afrag[ks], bb, acc[dt]);
    }

  bf16_t* dst = att + ((size_t)(b * 4096 + nbase)) * 1024 + h * 64;
#pragma unroll
  for (int dt = 0; dt < 4; dt++) {
#pragma unroll
    for (int e = 0; e < 8; e++) {
      int row = half * 8 + e;
      float nrm = __shfl(dot, row, 32) + 1e-6f;
      float val = acc[dt][e] / nrm;
      dst[(size_t)row * 1024 + dt * 16 + r] = f2bf(val);
    }
  }
}

// ---------------------------------------------------------------------------
// Workspace layout (bytes)
// ---------------------------------------------------------------------------
static const size_t OFF_QKV  = 0;                       // 16384*3072*2 = 96 MB
static const size_t OFF_QP   = 100663296;               // 32 MB
static const size_t OFF_KP   = 134217728;               // 32 MB
static const size_t OFF_KV   = 167772160;               // 1 MB
static const size_t OFF_KSUM = 168820736;               // 16 KB
static const size_t OFF_ATT  = 168837120;               // 32 MB

extern "C" void kernel_launch(void* const* d_in, const int* in_sizes, int n_in,
                              void* d_out, int out_size, void* d_ws, size_t ws_size,
                              hipStream_t stream) {
  const float* x        = (const float*)d_in[0];
  const float* w_qkv    = (const float*)d_in[1];
  const float* b_qkv    = (const float*)d_in[2];
  const float* w_proj   = (const float*)d_in[3];
  const float* b_proj   = (const float*)d_in[4];
  const float* proj_mat = (const float*)d_in[5];
  float* out = (float*)d_out;
  char* ws = (char*)d_ws;

  bf16_t* qkv = (bf16_t*)(ws + OFF_QKV);
  bf16_t* qp  = (bf16_t*)(ws + OFF_QP);
  bf16_t* kp  = (bf16_t*)(ws + OFF_KP);
  float*  kvb = (float*)(ws + OFF_KV);
  float*  ksm = (float*)(ws + OFF_KSUM);
  bf16_t* att = (bf16_t*)(ws + OFF_ATT);

  gemm_bf16_kernel<float, bf16_t><<<dim3(128, 24), 256, 0, stream>>>(
      x, w_qkv, b_qkv, qkv, 16384, 1024, 3072);
  feature_kernel<<<dim3(32, 2, 64), 256, 0, stream>>>(qkv, proj_mat, qp, kp);
  kv_kernel<<<dim3(64), 256, 0, stream>>>(qkv, kp, kvb, ksm);
  out_kernel<<<dim3(32, 64), 256, 0, stream>>>(qp, kvb, ksm, att);
  gemm_bf16_kernel<bf16_t, float><<<dim3(128, 8), 256, 0, stream>>>(
      att, w_proj, b_proj, out, 16384, 1024, 1024);
}
